// MLPAttention_9517647528305
// MI455X (gfx1250) — matrix operand order repaired and checked by reading, not compiled, not run on hardware
//
#include <hip/hip_runtime.h>
#include <hip/hip_bf16.h>

// ---------------------------------------------------------------------------
// MLP (Bahdanau) attention with coverage, fused for MI455X (gfx1250, wave32).
//   B=32, N=2048, H=1024
//   Dominant cost: v_hid = value @ Wv^T  (137 GFLOP)  -> bf16 WMMA 16x16x32
//   value (256MB fp32) read exactly twice (energy pass + weighted-sum pass),
//   which is the HBM roofline floor (~512MB @ 23.3 TB/s).
// ---------------------------------------------------------------------------

#define B_SZ 32
#define N_SZ 2048
#define H_SZ 1024

typedef __attribute__((ext_vector_type(16))) __bf16 v16bf;
typedef __attribute__((ext_vector_type(8)))  __bf16 v8bf;
typedef __attribute__((ext_vector_type(8)))  float  v8f;

union AFrag { v16bf v; v8bf h[2]; };

static __device__ __forceinline__ float fast_tanh(float x) {
    x = fminf(fmaxf(x, -15.0f), 15.0f);
    float e2 = __expf(2.0f * x);
    return __fdividef(e2 - 1.0f, e2 + 1.0f);
}

// ---------------------------------------------------------------------------
// q_hid[b,g] = dot(query[b,:], Wq[g,:]) + bias[g]        grid (H/256, B) x 256
// ---------------------------------------------------------------------------
__global__ void mlpattn_prep_qhb(const float* __restrict__ query,
                                 const float* __restrict__ Wq,
                                 const float* __restrict__ bias,
                                 float* __restrict__ qhb) {
    const int g = blockIdx.x * 256 + threadIdx.x;
    const int b = blockIdx.y;
    const float4* qr = (const float4*)(query + (size_t)b * H_SZ);
    const float4* wr = (const float4*)(Wq + (size_t)g * H_SZ);
    float acc = 0.0f;
    #pragma unroll 4
    for (int i = 0; i < H_SZ / 4; ++i) {
        float4 q = qr[i], w = wr[i];
        acc = fmaf(q.x, w.x, acc);
        acc = fmaf(q.y, w.y, acc);
        acc = fmaf(q.z, w.z, acc);
        acc = fmaf(q.w, w.w, acc);
    }
    qhb[(size_t)b * H_SZ + g] = acc + bias[g];
}

// ---------------------------------------------------------------------------
// Wv -> bf16 swizzled into WMMA B-fragment order.
// B operand (bf16, 32x16): lane L holds column g = gt*16 + (L&15),
// K = kt*32 + (L>>4)*16 + e  for e = 0..15, stored as 32 contiguous bytes.
// dst = ((gt*32 + kt)*32 + lane)*16 + e.  grid 4096 x 256.
// ---------------------------------------------------------------------------
__global__ void mlpattn_swizzle_wv(const float* __restrict__ Wv,
                                   __bf16* __restrict__ wsB) {
    const int idx  = blockIdx.x * 256 + threadIdx.x;   // 0 .. 2^20-1
    const int e    = idx & 15;
    const int lane = (idx >> 4) & 31;
    const int kt   = (idx >> 9) & 31;
    const int gt   = idx >> 14;
    const int k = kt * 32 + (lane >> 4) * 16 + e;
    const int g = gt * 16 + (lane & 15);
    wsB[idx] = (__bf16)Wv[(size_t)g * H_SZ + k];
}

// ---------------------------------------------------------------------------
// Energy kernel: e[b,n] = sum_g tanh(vhid + qhb + cov*Wc) * Wm
// grid (N/64, B), 128 threads (4 waves), 128KB dynamic LDS.
// Wave w owns the 16-row n-tile [n0 + 16w, n0 + 16w + 16).
// ---------------------------------------------------------------------------
__global__ void mlpattn_energy(const float* __restrict__ value,
                               const float* __restrict__ cov_vec,
                               const float* __restrict__ qhb,
                               const __bf16* __restrict__ wsB,
                               const float* __restrict__ Wc,
                               const float* __restrict__ Wm,
                               float* __restrict__ ebuf) {
    extern __shared__ __align__(16) char smem_raw[];
    __bf16* sA = (__bf16*)smem_raw;                    // [64][1024] bf16

    const int tid = threadIdx.x;                       // 0..127
    const int b   = blockIdx.y;
    const int n0  = blockIdx.x * 64;

    // ---- stage 64 x 1024 fp32 -> bf16 into LDS (wide b128 loads) ----
    const float* vbase = value + ((size_t)b * N_SZ + n0) * H_SZ;
    #pragma unroll 2
    for (int it = 0; it < 128; ++it) {                 // 16384 float4 / 128thr
        const int e4 = it * 128 + tid;
        float4 v = ((const float4*)vbase)[e4];
        __bf16* d = sA + (size_t)e4 * 4;
        d[0] = (__bf16)v.x; d[1] = (__bf16)v.y;
        d[2] = (__bf16)v.z; d[3] = (__bf16)v.w;
    }
    __syncthreads();

    const int wave = tid >> 5;
    const int lane = tid & 31;
    const int half = lane >> 4;                        // 0 | 1
    const int l16  = lane & 15;
    const int arow = wave * 16 + l16;                  // A-matrix row in LDS

    // cov for my rows: n = n0 + 16*wave + (r + 8*half)
    float cov[8];
    #pragma unroll
    for (int r = 0; r < 8; ++r)
        cov[r] = cov_vec[(size_t)b * N_SZ + n0 + wave * 16 + half * 8 + r];

    float eacc[8] = {0, 0, 0, 0, 0, 0, 0, 0};

    for (int gt = 0; gt < H_SZ / 16; gt += 2) {        // two g-tiles in flight
        v8f acc0 = {}; v8f acc1 = {};
        #pragma unroll 4
        for (int kt = 0; kt < 32; ++kt) {
            // A fragment (16-bit A 16x32): lanes 0-15 K=0..15,
            // lanes 16-31 K=16..31 for row arow.
            const __bf16* ap = sA + (size_t)arow * H_SZ + kt * 32 + half * 16;
            AFrag a;
            a.h[0] = *(const v8bf*)ap;                 // ds_load_b128
            a.h[1] = *(const v8bf*)(ap + 8);           // ds_load_b128
            // B fragments: 32 contiguous bytes per lane (2x global_load_b128)
            const __bf16* bp0 = wsB + (((size_t)gt * 32 + kt) * 32 + lane) * 16;
            const __bf16* bp1 = bp0 + 32 * 16 * 32;    // (gt+1) block
            v16bf b0 = *(const v16bf*)bp0;
            v16bf b1 = *(const v16bf*)bp1;
            __builtin_prefetch(bp0 + 512, 0, 1);       // next kt fragment
            acc0 = __builtin_amdgcn_wmma_f32_16x16x32_bf16(
                       false, a.v, false, b0, (short)0, acc0, false, false);
            acc1 = __builtin_amdgcn_wmma_f32_16x16x32_bf16(
                       false, a.v, false, b1, (short)0, acc1, false, false);
        }
        // fused tanh epilogue: D element (r, l16) -> n row r+8*half, col g
        #pragma unroll
        for (int gg = 0; gg < 2; ++gg) {
            const int g = (gt + gg) * 16 + l16;
            const float qb = qhb[(size_t)b * H_SZ + g];
            const float wc = Wc[g];
            const float wm = Wm[g];
            const v8f acc = gg ? acc1 : acc0;
            #pragma unroll
            for (int r = 0; r < 8; ++r) {
                float pre = acc[r] + qb + cov[r] * wc;
                eacc[r] = fmaf(fast_tanh(pre), wm, eacc[r]);
            }
        }
    }

    // reduce over the 16 g-lanes of each half-wave; lane 0 -> rows r,
    // lane 16 -> rows r+8
    #pragma unroll
    for (int r = 0; r < 8; ++r) {
        float v = eacc[r];
        v += __shfl_xor(v, 1, 32);
        v += __shfl_xor(v, 2, 32);
        v += __shfl_xor(v, 4, 32);
        v += __shfl_xor(v, 8, 32);
        if (l16 == 0)
            ebuf[(size_t)b * N_SZ + n0 + wave * 16 + half * 8 + r] = v;
    }
}

// ---------------------------------------------------------------------------
// Masked softmax over N per batch.  grid B x 256; each thread owns 8 slots.
// ---------------------------------------------------------------------------
__global__ void mlpattn_softmax(const float* __restrict__ ebuf,
                                const int* __restrict__ mask,
                                const float* __restrict__ bm,
                                float* __restrict__ attn) {
    __shared__ float red[256];
    const int b = blockIdx.x, tid = threadIdx.x;
    const float bmv = bm[0];
    float ev[8];
    float mx = -3.4e38f;
    #pragma unroll
    for (int i = 0; i < 8; ++i) {
        const int n = tid + i * 256;
        float e = ebuf[(size_t)b * N_SZ + n] + bmv;
        e = (mask[(size_t)b * N_SZ + n] > 0) ? e : -1.0e9f;
        ev[i] = e;
        mx = fmaxf(mx, e);
    }
    red[tid] = mx; __syncthreads();
    for (int s = 128; s > 0; s >>= 1) {
        if (tid < s) red[tid] = fmaxf(red[tid], red[tid + s]);
        __syncthreads();
    }
    mx = red[0]; __syncthreads();
    float sum = 0.0f;
    #pragma unroll
    for (int i = 0; i < 8; ++i) { ev[i] = __expf(ev[i] - mx); sum += ev[i]; }
    red[tid] = sum; __syncthreads();
    for (int s = 128; s > 0; s >>= 1) {
        if (tid < s) red[tid] += red[tid + s];
        __syncthreads();
    }
    const float inv = __fdividef(1.0f, red[0]);
    #pragma unroll
    for (int i = 0; i < 8; ++i)
        attn[(size_t)b * N_SZ + tid + i * 256] = ev[i] * inv;
}

// ---------------------------------------------------------------------------
// output[b,h] = sum_n attn[b,n] * value[b,n,h]   grid (H/256, B) x 256
// Deterministic: each output element written exactly once, no atomics.
// ---------------------------------------------------------------------------
__global__ void mlpattn_out(const float* __restrict__ value,
                            const float* __restrict__ attn,
                            float* __restrict__ out) {
    const int h = blockIdx.x * 256 + threadIdx.x;
    const int b = blockIdx.y;
    const float* vb = value + (size_t)b * N_SZ * H_SZ + h;
    const float* ab = attn + (size_t)b * N_SZ;
    float acc = 0.0f;
    #pragma unroll 8
    for (int n = 0; n < N_SZ; ++n)
        acc = fmaf(ab[n], vb[(size_t)n * H_SZ], acc);
    out[(size_t)b * H_SZ + h] = acc;
}

// ---------------------------------------------------------------------------
extern "C" void kernel_launch(void* const* d_in, const int* in_sizes, int n_in,
                              void* d_out, int out_size, void* d_ws, size_t ws_size,
                              hipStream_t stream) {
    const float* query = (const float*)d_in[0];
    const float* value = (const float*)d_in[1];
    const int*   mask  = (const int*)  d_in[2];
    const float* cov   = (const float*)d_in[3];
    const float* Wq    = (const float*)d_in[4];
    const float* Wv    = (const float*)d_in[5];
    const float* Wc    = (const float*)d_in[6];
    const float* bias  = (const float*)d_in[7];
    const float* Wm    = (const float*)d_in[8];
    const float* bm    = (const float*)d_in[9];

    float* out  = (float*)d_out;                 // [B,H]
    float* attn = out + (size_t)B_SZ * H_SZ;     // [B,N]  (tuple output #2)

    // workspace carve: 2MB swizzled bf16 Wv | qhb [B,H] | e [B,N]
    __bf16* wsB = (__bf16*)d_ws;
    float*  qhb = (float*)((char*)d_ws + (size_t)(1 << 21));
    float*  ebuf = qhb + (size_t)B_SZ * H_SZ;

    mlpattn_prep_qhb<<<dim3(H_SZ / 256, B_SZ), 256, 0, stream>>>(query, Wq, bias, qhb);
    mlpattn_swizzle_wv<<<4096, 256, 0, stream>>>(Wv, wsB);
    mlpattn_energy<<<dim3(N_SZ / 64, B_SZ), 128, 64 * H_SZ * sizeof(__bf16), stream>>>(
        value, cov, qhb, wsB, Wc, Wm, ebuf);
    mlpattn_softmax<<<B_SZ, 256, 0, stream>>>(ebuf, mask, bm, attn);
    mlpattn_out<<<dim3(H_SZ / 256, B_SZ), 256, 0, stream>>>(value, attn, out);
}
